// DecoderStackLSTM_34110630265351
// MI455X (gfx1250) — matrix operand order repaired and checked by reading, not compiled, not run on hardware
//
#include <hip/hip_runtime.h>
#include <hip/hip_bf16.h>

// ---------------- problem constants ----------------
#define S_LEN 128
#define BATCH 64
#define EMB   512
#define HID   1024
#define GATES (4 * HID)            // 4096
#define K0    (EMB + HID + HID)    // 2560 (x = [emb | out] ++ h0)
#define K1    (HID + HID)          // 2048 ([h0_new | h1])
#define NKT0  (K0 / 32)            // 80 k-tiles
#define NKT1  (K1 / 32)            // 64 k-tiles
#define NWG   (HID / 16)           // 64 workgroups, 16 h-columns each
#define NTHR  128                  // 4 waves -> 4 M-tiles of 16 (M = BATCH = 64)
#define W1_LDS_BYTES (GATES / 4 * 16 * K1 * 2 / (HID / 16)) // = 1 MiB; per-WG W1 fragments use 4*64KiB = 256KiB

typedef __attribute__((ext_vector_type(16))) __bf16 v16bf;
typedef __attribute__((ext_vector_type(8)))  __bf16 v8bf;
typedef __attribute__((ext_vector_type(8)))  float  v8f;

// ---------------- device helpers ----------------
__device__ __forceinline__ float sigf(float x) {
    return 1.0f / (1.0f + __expf(-x));
}

// Layer-0 GEMM segment: B streamed from global (L2-resident), 3-stage register
// pipeline with prefetch distance 2 so global_load_b128 overlaps v_wmma.
// A row-major bf16 [BATCH, ldA]; W packed WMMA-B fragments:
// tile (nt, kt) -> 512 bf16 at ((nt*nktiles + kt) << 9), 32B contiguous per lane.
template<int NK>
__device__ __forceinline__ void gemm_seg_pipe(v8f acc[4],
                                              const __bf16* __restrict__ A, int ldA,
                                              int kbase,
                                              const __bf16* __restrict__ Wp, int nktiles,
                                              int wg, int lane, int arow) {
    const int hs8 = (lane >> 4) << 3;                 // 0 or 8 (K half-select)
    const __bf16* arowp = A + (size_t)arow * ldA + hs8;
    const __bf16* wb[4];
#pragma unroll
    for (int g = 0; g < 4; ++g)
        wb[g] = Wp + (((size_t)((g * NWG + wg) * nktiles + (kbase >> 5))) << 9)
                   + (lane << 4);

    v16bf af[3];
    v16bf bfr[3][4];

    auto LOAD = [&](int s, int k) {
        v8bf alo = *(const v8bf*)(arowp + (k << 5));
        v8bf ahi = *(const v8bf*)(arowp + (k << 5) + 16);
        af[s] = __builtin_shufflevector(alo, ahi,
                    0, 1, 2, 3, 4, 5, 6, 7, 8, 9, 10, 11, 12, 13, 14, 15);
#pragma unroll
        for (int g = 0; g < 4; ++g)
            bfr[s][g] = *(const v16bf*)(wb[g] + ((size_t)k << 9));
    };

    LOAD(0, 0);
    LOAD(1, 1);
#pragma unroll
    for (int k = 0; k < NK; ++k) {
        const int cs = k % 3;
        if (k + 2 < NK) LOAD((k + 2) % 3, k + 2);
#pragma unroll
        for (int g = 0; g < 4; ++g)
            acc[g] = __builtin_amdgcn_wmma_f32_16x16x32_bf16(
                false, af[cs], false, bfr[cs][g], (short)0, acc[g], false, false);
    }
}

// Layer-1 GEMM segment: B fragments read from LDS-resident W1 (ds_load),
// A prefetched 1 tile ahead from global.
template<int NK>
__device__ __forceinline__ void gemm_seg_lds(v8f acc[4],
                                             const __bf16* __restrict__ A, int ldA,
                                             int ktbase, int lane, int arow) {
    extern __shared__ __bf16 smem[];                  // W1 fragments, 256 KiB
    const int hs8 = (lane >> 4) << 3;
    const __bf16* arowp = A + (size_t)arow * ldA + hs8;

    v16bf af[2];
    auto LOADA = [&](int s, int k) {
        v8bf alo = *(const v8bf*)(arowp + (k << 5));
        v8bf ahi = *(const v8bf*)(arowp + (k << 5) + 16);
        af[s] = __builtin_shufflevector(alo, ahi,
                    0, 1, 2, 3, 4, 5, 6, 7, 8, 9, 10, 11, 12, 13, 14, 15);
    };

    LOADA(0, 0);
#pragma unroll
    for (int k = 0; k < NK; ++k) {
        if (k + 1 < NK) LOADA((k + 1) & 1, k + 1);
#pragma unroll
        for (int g = 0; g < 4; ++g) {
            const v16bf bfrag = *(const v16bf*)
                (smem + (((g * NKT1 + ktbase + k) << 9) + (lane << 4)));
            acc[g] = __builtin_amdgcn_wmma_f32_16x16x32_bf16(
                false, af[k & 1], false, bfrag, (short)0, acc[g], false, false);
        }
    }
}

// Device-wide arrive + spin barrier (per-phase counters, zeroed each call).
__device__ __forceinline__ void grid_barrier(unsigned* ctr, unsigned nwg) {
    __syncthreads();
    if (threadIdx.x == 0) {
        __threadfence();
        __hip_atomic_fetch_add(ctr, 1u, __ATOMIC_RELEASE, __HIP_MEMORY_SCOPE_AGENT);
        while (__hip_atomic_load(ctr, __ATOMIC_ACQUIRE, __HIP_MEMORY_SCOPE_AGENT) < nwg) {
            __builtin_amdgcn_s_sleep(8);
        }
    }
    __syncthreads();
}

// ---------------- prep kernels ----------------
__global__ void k_embed(const int* __restrict__ tgt, const float* __restrict__ embW,
                        __bf16* __restrict__ out) {
    int idx = blockIdx.x * blockDim.x + threadIdx.x;   // S*B*E total
    int e  = idx & (EMB - 1);
    int sb = idx >> 9;
    int tok = tgt[sb];
    out[idx] = (__bf16)embW[(size_t)tok * EMB + e];
}

__global__ void k_pack(const float* __restrict__ Wi, const float* __restrict__ Wh,
                       __bf16* __restrict__ dst, int Kx, int Ktot) {
    int idx  = blockIdx.x * blockDim.x + threadIdx.x;  // GATES*Ktot total
    int tile = idx >> 9;
    int r    = idx & 511;
    int lane = r >> 4;
    int j    = r & 15;
    int nkt  = Ktot >> 5;
    int nt   = tile / nkt;
    int kt   = tile - nt * nkt;
    int n    = (nt << 4) + (lane & 15);
    int k    = (kt << 5) + ((lane >> 4) << 3) + (j & 7) + ((j >> 3) << 4);
    float v  = (k < Kx) ? Wi[(size_t)n * Kx + k]
                        : Wh[(size_t)n * HID + (k - Kx)];
    dst[idx] = (__bf16)v;
}

__global__ void k_init(const float* __restrict__ h0, const float* __restrict__ c0,
                       const float* __restrict__ initout,
                       __bf16* __restrict__ h0a, __bf16* __restrict__ h1a,
                       __bf16* __restrict__ outbf,
                       float* __restrict__ c0st, float* __restrict__ c1st) {
    int i = blockIdx.x * blockDim.x + threadIdx.x;     // B*H total
    h0a[i]   = (__bf16)h0[i];
    h1a[i]   = (__bf16)h0[BATCH * HID + i];
    outbf[i] = (__bf16)initout[i];
    c0st[i]  = c0[i];
    c1st[i]  = c0[BATCH * HID + i];
}

// ---------------- persistent recurrent kernel ----------------
struct LstmParams {
    const __bf16* embs;        // [S, B, E] bf16
    const __bf16* w0;          // packed [4096, 2560]
    const __bf16* w1;          // packed [4096, 2048]
    __bf16* h0buf0; __bf16* h0buf1;   // double-buffered h0 (bf16 [B,H])
    __bf16* h1buf0; __bf16* h1buf1;   // double-buffered h1 (bf16 [B,H])
    const __bf16* initout;     // bf16 [B,H]
    float* c0st; float* c1st;  // fp32 cell state [B,H]
    const float* bih0; const float* bhh0;
    const float* bih1; const float* bhh1;
    float* outputs;            // [S, B, H]
    float* hn0; float* hn1;    // d_out h_n slots
    float* cn0; float* cn1;    // d_out c_n slots
    unsigned* bar;             // 2*S phase counters
};

__global__ __launch_bounds__(NTHR, 1) void k_lstm_persistent(LstmParams p) {
    extern __shared__ __bf16 smem[];         // 256 KiB: this WG's W1 fragments
    const int wg   = blockIdx.x;             // 0..63 : h columns [16*wg, 16*wg+16)
    const int lane = threadIdx.x & 31;
    const int wave = threadIdx.x >> 5;       // 0..3  : batch rows [16*wave, ..+16)
    const int n0   = wg << 4;
    const int m0   = wave << 4;
    const int col  = n0 + (lane & 15);
    const int arow = m0 + (lane & 15);
    const int roff = (lane >> 4) << 3;       // +8 row offset for lanes 16..31

    // ---- one-time: async-copy this WG's W1 fragments (4 x 64 KiB) into LDS ----
    {
        const unsigned lds0 = (unsigned)(uintptr_t)smem;
        const int tid = threadIdx.x;
#pragma unroll 1
        for (int g = 0; g < 4; ++g) {
            const __bf16* src = p.w1 + (((size_t)(g * NWG + wg) * NKT1) << 9);
            unsigned long long sbase = (unsigned long long)(uintptr_t)src;
#pragma unroll 1
            for (int i = 0; i < 32; ++i) {             // 32 * 128thr * 16B = 64 KiB
                unsigned boff = (unsigned)((i * NTHR + tid) << 4);
                unsigned ldsa = lds0 + ((unsigned)g << 16) + boff;
                asm volatile("global_load_async_to_lds_b128 %0, %1, %2"
                             :: "v"(ldsa), "v"(boff), "s"(sbase) : "memory");
            }
        }
        asm volatile("s_wait_asynccnt 0x0" ::: "memory");
        __syncthreads();
    }

    // per-gate biases for this column (i, f, g, o)
    float b0[4], b1[4];
#pragma unroll
    for (int g = 0; g < 4; ++g) {
        b0[g] = p.bih0[g * HID + col] + p.bhh0[g * HID + col];
        b1[g] = p.bih1[g * HID + col] + p.bhh1[g * HID + col];
    }

    __bf16* h0b[2] = { p.h0buf0, p.h0buf1 };
    __bf16* h1b[2] = { p.h1buf0, p.h1buf1 };
    int cur = 0;

    for (int t = 0; t < S_LEN; ++t) {
        // ---------------- layer 0 (B streamed from L2, pipelined) ----------------
        {
            const __bf16* embt   = p.embs + (size_t)t * BATCH * EMB;
            const __bf16* outsrc = (t == 0) ? p.initout : h1b[cur];
            const __bf16* h0src  = h0b[cur];

            v8f acc[4] = {};
            gemm_seg_pipe<16>(acc, embt,   EMB,       0, p.w0, NKT0, wg, lane, arow);
            gemm_seg_pipe<32>(acc, outsrc, HID,     EMB, p.w0, NKT0, wg, lane, arow);
            gemm_seg_pipe<32>(acc, h0src,  HID, EMB+HID, p.w0, NKT0, wg, lane, arow);

            __bf16* h0dst = h0b[cur ^ 1];
#pragma unroll
            for (int r = 0; r < 8; ++r) {
                int row = m0 + r + roff;
                float xi = acc[0][r] + b0[0];
                float xf = acc[1][r] + b0[1];
                float xg = acc[2][r] + b0[2];
                float xo = acc[3][r] + b0[3];
                size_t o = (size_t)row * HID + col;
                float cp = p.c0st[o];
                float cn = sigf(xf) * cp + sigf(xi) * tanhf(xg);
                float hn = sigf(xo) * tanhf(cn);
                p.c0st[o] = cn;
                p.cn0[o]  = cn;
                p.hn0[o]  = hn;
                h0dst[o]  = (__bf16)hn;
            }
        }
        grid_barrier(p.bar + 2 * t, NWG);

        // ---------------- layer 1 (B LDS-resident) ----------------
        {
            const __bf16* h0new = h0b[cur ^ 1];
            const __bf16* h1src = h1b[cur];

            v8f acc[4] = {};
            gemm_seg_lds<32>(acc, h0new, HID,  0, lane, arow);
            gemm_seg_lds<32>(acc, h1src, HID, 32, lane, arow);

            __bf16* h1dst = h1b[cur ^ 1];
            float* outt = p.outputs + (size_t)t * BATCH * HID;
#pragma unroll
            for (int r = 0; r < 8; ++r) {
                int row = m0 + r + roff;
                float xi = acc[0][r] + b1[0];
                float xf = acc[1][r] + b1[1];
                float xg = acc[2][r] + b1[2];
                float xo = acc[3][r] + b1[3];
                size_t o = (size_t)row * HID + col;
                float cp = p.c1st[o];
                float cn = sigf(xf) * cp + sigf(xi) * tanhf(xg);
                float hn = sigf(xo) * tanhf(cn);
                p.c1st[o] = cn;
                p.cn1[o]  = cn;
                p.hn1[o]  = hn;
                outt[o]   = hn;
                h1dst[o]  = (__bf16)hn;
            }
        }
        grid_barrier(p.bar + 2 * t + 1, NWG);

        cur ^= 1;
    }
}

// ---------------- host launcher ----------------
extern "C" void kernel_launch(void* const* d_in, const int* in_sizes, int n_in,
                              void* d_out, int out_size, void* d_ws, size_t ws_size,
                              hipStream_t stream) {
    (void)in_sizes; (void)n_in; (void)out_size; (void)ws_size;

    const int*   tgt      = (const int*)  d_in[0];
    const float* h0       = (const float*)d_in[1];
    const float* c0       = (const float*)d_in[2];
    const float* initout  = (const float*)d_in[3];
    const float* embW     = (const float*)d_in[4];
    const float* W_ih0    = (const float*)d_in[5];
    const float* W_hh0    = (const float*)d_in[6];
    const float* b_ih0    = (const float*)d_in[7];
    const float* b_hh0    = (const float*)d_in[8];
    const float* W_ih1    = (const float*)d_in[9];
    const float* W_hh1    = (const float*)d_in[10];
    const float* b_ih1    = (const float*)d_in[11];
    const float* b_hh1    = (const float*)d_in[12];

    char* ws = (char*)d_ws;
    size_t off = 0;
    auto carve = [&](size_t bytes) {
        char* ptr = ws + off;
        off = (off + bytes + 255) & ~(size_t)255;
        return ptr;
    };
    __bf16* embs  = (__bf16*)carve((size_t)S_LEN * BATCH * EMB * 2);
    __bf16* w0p   = (__bf16*)carve((size_t)GATES * K0 * 2);
    __bf16* w1p   = (__bf16*)carve((size_t)GATES * K1 * 2);
    __bf16* h0a   = (__bf16*)carve((size_t)BATCH * HID * 2);
    __bf16* h0bb  = (__bf16*)carve((size_t)BATCH * HID * 2);
    __bf16* h1a   = (__bf16*)carve((size_t)BATCH * HID * 2);
    __bf16* h1bb  = (__bf16*)carve((size_t)BATCH * HID * 2);
    __bf16* outbf = (__bf16*)carve((size_t)BATCH * HID * 2);
    float*  c0st  = (float*) carve((size_t)BATCH * HID * 4);
    float*  c1st  = (float*) carve((size_t)BATCH * HID * 4);
    unsigned* bar = (unsigned*)carve((size_t)2 * S_LEN * 4);

    // d_out layout: outputs [S,B,H] ++ h_n [2,B,H] ++ c_n [2,B,H]
    float* out     = (float*)d_out;
    float* outputs = out;
    float* hn0 = out + (size_t)S_LEN * BATCH * HID;
    float* hn1 = hn0 + (size_t)BATCH * HID;
    float* cn0 = hn1 + (size_t)BATCH * HID;
    float* cn1 = cn0 + (size_t)BATCH * HID;

    hipMemsetAsync(bar, 0, (size_t)2 * S_LEN * 4, stream);

    k_embed<<<(S_LEN * BATCH * EMB) / 256, 256, 0, stream>>>(tgt, embW, embs);
    k_pack <<<((size_t)GATES * K0) / 256, 256, 0, stream>>>(W_ih0, W_hh0, w0p, EMB + HID, K0);
    k_pack <<<((size_t)GATES * K1) / 256, 256, 0, stream>>>(W_ih1, W_hh1, w1p, HID, K1);
    k_init <<<(BATCH * HID) / 256, 256, 0, stream>>>(h0, c0, initout, h0a, h1a, outbf, c0st, c1st);

    LstmParams p;
    p.embs = embs; p.w0 = w0p; p.w1 = w1p;
    p.h0buf0 = h0a; p.h0buf1 = h0bb;
    p.h1buf0 = h1a; p.h1buf1 = h1bb;
    p.initout = outbf;
    p.c0st = c0st; p.c1st = c1st;
    p.bih0 = b_ih0; p.bhh0 = b_hh0; p.bih1 = b_ih1; p.bhh1 = b_hh1;
    p.outputs = outputs; p.hn0 = hn0; p.hn1 = hn1; p.cn0 = cn0; p.cn1 = cn1;
    p.bar = bar;

    // 1 MiB dynamic LDS requested for the 256 KiB of W1-resident tiles
    hipFuncSetAttribute((const void*)k_lstm_persistent,
                        hipFuncAttributeMaxDynamicSharedMemorySize, W1_LDS_BYTES);
    k_lstm_persistent<<<dim3(NWG), dim3(NTHR), W1_LDS_BYTES, stream>>>(p);
}
